// Network_18717467476537
// MI455X (gfx1250) — matrix operand order repaired and checked by reading, not compiled, not run on hardware
//
#include <hip/hip_runtime.h>

// ---------------- problem constants ----------------
#define BATCH 2048
#define F0    39
#define EMB   16
#define LCR   128      // L
#define HCR   64       // H
#define DXC   13
#define DIN   829      // DX + F0*EMB + (H+L)
#define ROWS  (BATCH*EMB)   // 32768 (b,d) rows
#define K1R   1521          // F0*F0
#define K1P   1536          // padded to mult of 32
#define K2R   2496          // F0*H (already mult of 32)

typedef __attribute__((ext_vector_type(16))) __bf16 v16bf;
typedef __attribute__((ext_vector_type(8)))  float  v8f;

// LDS layout inside cross_gemm (bytes from dynamic-LDS base):
//   Xl  (bf16 [128][40])          : 0     .. 10240
//   WlA (bf16 swizzled 32x128)    : 10240 .. 18432
//   WlB (bf16 swizzled 32x128)    : 18432 .. 26624
//   Cl  (bf16 [128][64], MODE 2)  : 26624 .. 43008
#define XL_OFF  0
#define WLA_OFF 10240
#define WLB_OFF 18432
#define CL_OFF  26624

// ---------------- gather + layout kernel ----------------
__global__ __launch_bounds__(256) void gather_kernel(
    const int* __restrict__ hash, const float* __restrict__ table,
    const float* __restrict__ input_x,
    __bf16* __restrict__ Xr, float* __restrict__ inputs_raw)
{
    int gid = blockIdx.x * 256 + threadIdx.x;      // over BATCH*F0 = 79872
    if (gid >= BATCH * F0) return;
    int b = gid / F0;
    int i = gid - b * F0;
    size_t idx = (size_t)hash[gid];
    const float4* src = (const float4*)(table + idx * 16);
    float4 e0 = src[0], e1 = src[1], e2 = src[2], e3 = src[3];
    float v[16] = { e0.x,e0.y,e0.z,e0.w, e1.x,e1.y,e1.z,e1.w,
                    e2.x,e2.y,e2.z,e2.w, e3.x,e3.y,e3.z,e3.w };
    float4* dst = (float4*)(inputs_raw + (size_t)b * DIN + DXC + i * 16);
    dst[0] = e0; dst[1] = e1; dst[2] = e2; dst[3] = e3;
    #pragma unroll
    for (int d = 0; d < 16; ++d)
        Xr[(size_t)(b * 16 + d) * 40 + i] = (__bf16)v[d];
    if (i == 0) {
        #pragma unroll
        for (int d = 0; d < 16; ++d)
            Xr[(size_t)(b * 16 + d) * 40 + 39] = (__bf16)0.f;   // pad col
    }
    if (i < DXC)
        inputs_raw[(size_t)b * DIN + i] = input_x[(size_t)b * DXC + i];
}

// ---------------- f32 -> bf16 weight pack (zero-pad K rows) ----------------
__global__ __launch_bounds__(256) void wconv_kernel(
    const float* __restrict__ w, __bf16* __restrict__ o,
    int total_real, int total_pad)
{
    int g = blockIdx.x * 256 + threadIdx.x;
    if (g < total_pad)
        o[g] = (g < total_real) ? (__bf16)w[g] : (__bf16)0.f;
}

// ---------------- async W-chunk stage: global -> LDS, no VGPR round trip ----
// Each of 256 threads moves 32B: W[k0+r][nt*16 .. +15] -> swizzled slot so a
// B-fragment (K=lane, N=nt*16..+15) is 32B contiguous in LDS.
__device__ __forceinline__ void async_w_chunk(
    const __bf16* __restrict__ Wb, int k0, unsigned lds_dst_base, int tid)
{
    const int nt = tid & 7;
    const int r  = tid >> 3;                       // 0..31 (K within chunk)
    unsigned dst = lds_dst_base + (unsigned)((nt * 32 + r) * 32);
    unsigned long long src =
        (unsigned long long)(uintptr_t)&Wb[(size_t)(k0 + r) * 128 + nt * 16];
    asm volatile("global_load_async_to_lds_b128 %0, %1, off\n\t"
                 "global_load_async_to_lds_b128 %0, %1, off offset:16"
                 :: "v"(dst), "v"(src) : "memory");
}

// ---------------- fused cross-layer WMMA GEMM ----------------
// MODE 1: C1 = relu(A1 x W0bf); cols 0..63 -> NH (next_h, bf16),
//         cols 64..127 -> wave-reduced over d into inputs_raw[., 637+m]
// MODE 2: C2 = relu(A2 x W1bf); all cols wave-reduced into inputs_raw[., 701+l]
// WG: 256 thr = 8 waves; 128 rows x 128 cols; wave w owns batch element
// b = blockIdx*8 + w (its 16 d-rows) -> deterministic in-wave d-reduction.
template<int MODE>
__global__ __launch_bounds__(256) void cross_gemm(
    const __bf16* __restrict__ Xr,       // [ROWS][40] bf16
    const __bf16* __restrict__ NH,       // [ROWS][64] bf16 (MODE 2 input)
    const __bf16* __restrict__ Wb,       // [KP][128] bf16
    __bf16* __restrict__ NHout,          // [ROWS][64] bf16 (MODE 1 output)
    float* __restrict__ inputs_raw,      // [BATCH][DIN]
    int KP, int Kreal)
{
    extern __shared__ char smem[];
    __bf16* Xl  = (__bf16*)(smem + XL_OFF);
    __bf16* WlA = (__bf16*)(smem + WLA_OFF);
    __bf16* WlB = (__bf16*)(smem + WLB_OFF);
    __bf16* Cl  = (__bf16*)(smem + CL_OFF);
    const unsigned ldsBase = (unsigned)(uintptr_t)smem;   // LDS byte offset

    const int tid   = threadIdx.x;
    const int wave  = tid >> 5;
    const int lane  = tid & 31;
    const int h     = lane >> 4;       // WMMA lane half
    const int mlane = lane & 15;
    const int rowbase = blockIdx.x * 128;

    // stage X rows (and next_h rows for MODE 2) once, as dwords
    {
        const unsigned* gx = (const unsigned*)(Xr + (size_t)rowbase * 40);
        unsigned* lx = (unsigned*)Xl;
        for (int idx = tid; idx < 128 * 40 / 2; idx += 256) lx[idx] = gx[idx];
        if (MODE == 2) {
            const unsigned* gc = (const unsigned*)(NH + (size_t)rowbase * 64);
            unsigned* lc = (unsigned*)Cl;
            for (int idx = tid; idx < 128 * 64 / 2; idx += 256) lc[idx] = gc[idx];
        }
    }

    const __bf16* xrow = &Xl[(wave * 16 + mlane) * 40];
    const __bf16* crow = &Cl[(wave * 16 + mlane) * 64];

    // kick off first W chunk via async DMA to LDS (buffer A)
    async_w_chunk(Wb, 0, ldsBase + WLA_OFF, tid);
    __syncthreads();   // X/C staging visible (async waited before use below)

    v8f acc[8] = {};
    const int nchunks = KP >> 5;
    for (int kc = 0; kc < nchunks; ++kc) {
        const int k0 = kc << 5;
        const bool hasNext = (kc + 1 < nchunks);
        // issue next chunk's DMA into the other buffer (its last reader
        // finished at the end-of-iteration barrier of kc-1)
        if (hasNext)
            async_w_chunk(Wb, k0 + 32,
                          ldsBase + (((kc + 1) & 1) ? WLB_OFF : WLA_OFF), tid);
        if (kc + 2 < nchunks && lane == 0)
            __builtin_prefetch(&Wb[(size_t)(k0 + 64) * 128 + wave * 16], 0, 1);

        // Build A fragment while DMA is in flight (depends only on Xl/Cl)
        v16bf afrag;
        #pragma unroll
        for (int v = 0; v < 8; ++v) {
            const int kb = (v < 4) ? (2 * v) : (16 + 2 * (v - 4));
            #pragma unroll
            for (int e = 0; e < 2; ++e) {
                const int c = k0 + kb + 8 * h + e;      // global K index
                float a;
                if (MODE == 1) {
                    int i = c / F0, j = c - i * F0;
                    a = (c < Kreal) ? (float)xrow[i] * (float)xrow[j] : 0.f;
                } else {
                    a = (float)xrow[c >> 6] * (float)crow[c & 63];
                }
                afrag[2 * v + e] = (__bf16)a;
            }
        }

        // current chunk's 2 async ops done (in-order completion); next 2 fly on
        if (hasNext) asm volatile("s_wait_asynccnt 0x2" ::: "memory");
        else         asm volatile("s_wait_asynccnt 0x0" ::: "memory");
        __syncthreads();

        const __bf16* Wcur = (kc & 1) ? WlB : WlA;
        #pragma unroll
        for (int g = 0; g < 2; ++g) {
            v16bf bf[4];
            #pragma unroll
            for (int q = 0; q < 4; ++q) {
                const __bf16* wp = &Wcur[(g * 4 + q) * 32 * 16 + mlane];
                #pragma unroll
                for (int s = 0; s < 16; ++s) {
                    const int kk = (s < 8) ? (8 * h + s) : (16 + 8 * h + (s - 8));
                    bf[q][s] = wp[kk * 16];
                }
            }
            #pragma unroll
            for (int q = 0; q < 4; ++q)
                acc[g * 4 + q] = __builtin_amdgcn_wmma_f32_16x16x32_bf16(
                    false, afrag, false, bf[q], (short)0, acc[g * 4 + q],
                    false, false);
        }
        __syncthreads();   // all reads of Wcur done before it is overwritten
    }

    const int wrow0 = rowbase + wave * 16;
    const int bidx  = blockIdx.x * 8 + wave;            // batch element
    if (MODE == 1) {
        #pragma unroll
        for (int nt = 0; nt < 4; ++nt) {                // next_h: cols 0..63
            #pragma unroll
            for (int r = 0; r < 8; ++r) {
                float vv = acc[nt][r]; vv = vv > 0.f ? vv : 0.f;
                NHout[(size_t)(wrow0 + r + 8 * h) * 64 + nt * 16 + mlane] = (__bf16)vv;
            }
        }
        #pragma unroll
        for (int nt = 4; nt < 8; ++nt) {                // direct0 -> sum over d
            float s = 0.f;
            #pragma unroll
            for (int r = 0; r < 8; ++r) { float vv = acc[nt][r]; s += vv > 0.f ? vv : 0.f; }
            s += __shfl_xor(s, 16, 32);
            if (h == 0)
                inputs_raw[(size_t)bidx * DIN + (DXC + F0 * EMB) + (nt - 4) * 16 + mlane] = s;
        }
    } else {
        #pragma unroll
        for (int nt = 0; nt < 8; ++nt) {                // direct1 -> sum over d
            float s = 0.f;
            #pragma unroll
            for (int r = 0; r < 8; ++r) { float vv = acc[nt][r]; s += vv > 0.f ? vv : 0.f; }
            s += __shfl_xor(s, 16, 32);
            if (h == 0)
                inputs_raw[(size_t)bidx * DIN + (DXC + F0 * EMB) + HCR + nt * 16 + mlane] = s;
        }
    }
}

// ---------------- batch-norm (train) over batch dim ----------------
__global__ __launch_bounds__(256) void bn_kernel(
    const float* __restrict__ x, float* __restrict__ y,
    const float* __restrict__ gamma, const float* __restrict__ beta,
    int C, int do_relu)
{
    __shared__ float sm[8][32], sm2[8][32];
    const int tc = threadIdx.x & 31, tr = threadIdx.x >> 5;
    const int col = blockIdx.x * 32 + tc;
    const bool valid = col < C;
    float s = 0.f, s2 = 0.f;
    if (valid)
        for (int r = tr; r < BATCH; r += 8) {
            float v = x[(size_t)r * C + col];
            s += v; s2 += v * v;
        }
    sm[tr][tc] = s; sm2[tr][tc] = s2;
    __syncthreads();
    if (tr == 0) {
        float a = 0.f, b = 0.f;
        #pragma unroll
        for (int k = 0; k < 8; ++k) { a += sm[k][tc]; b += sm2[k][tc]; }
        float mean = a * (1.f / BATCH);
        float var  = b * (1.f / BATCH) - mean * mean;
        sm[0][tc]  = mean;
        sm2[0][tc] = rsqrtf(var + 0.001f);
    }
    __syncthreads();
    if (valid) {
        const float mean = sm[0][tc], rstd = sm2[0][tc];
        const float g = gamma[col], be = beta[col];
        for (int r = tr; r < BATCH; r += 8) {
            float v = (x[(size_t)r * C + col] - mean) * rstd * g + be;
            if (do_relu) v = v > 0.f ? v : 0.f;
            y[(size_t)r * C + col] = v;
        }
    }
}

// ---------------- small MLP tail ----------------
__global__ __launch_bounds__(256) void mlp1_kernel(     // (B,829)x(829,64)+b3
    const float* __restrict__ xin, const float* __restrict__ w,
    const float* __restrict__ b, float* __restrict__ z)
{
    int g = blockIdx.x * 256 + threadIdx.x;             // over B*64
    int row = g >> 6, col = g & 63;
    float acc = b[col];
    const float* xr = xin + (size_t)row * DIN;
    for (int k = 0; k < DIN; ++k) acc += xr[k] * w[k * 64 + col];
    z[g] = acc;
}

__global__ __launch_bounds__(256) void mlp2_kernel(     // (B,64)x(64,32)+b4
    const float* __restrict__ xin, const float* __restrict__ w,
    const float* __restrict__ b, float* __restrict__ z)
{
    int g = blockIdx.x * 256 + threadIdx.x;             // over B*32
    int row = g >> 5, col = g & 31;
    float acc = b[col];
    const float* xr = xin + (size_t)row * 64;
    #pragma unroll
    for (int k = 0; k < 64; ++k) acc += xr[k] * w[k * 32 + col];
    z[g] = acc;
}

__global__ __launch_bounds__(256) void final_kernel(    // (B,32)x(32,1)+b5
    const float* __restrict__ h4, const float* __restrict__ w5,
    const float* __restrict__ b5, float* __restrict__ out)
{
    int row = blockIdx.x * 256 + threadIdx.x;           // over B
    if (row >= BATCH) return;
    float acc = b5[0];
    const float* xr = h4 + (size_t)row * 32;
    #pragma unroll
    for (int k = 0; k < 32; ++k) acc += xr[k] * w5[k];
    out[row] = acc;
}

// ---------------- launch ----------------
extern "C" void kernel_launch(void* const* d_in, const int* in_sizes, int n_in,
                              void* d_out, int out_size, void* d_ws, size_t ws_size,
                              hipStream_t stream)
{
    (void)in_sizes; (void)n_in; (void)out_size; (void)ws_size;
    const float* input_x  = (const float*)d_in[0];
    const int*   in_hash  = (const int*)  d_in[1];
    const float* table    = (const float*)d_in[2];
    const float* W0       = (const float*)d_in[3];
    const float* W1       = (const float*)d_in[4];
    const float* gamma1   = (const float*)d_in[5];
    const float* beta1    = (const float*)d_in[6];
    const float* w3       = (const float*)d_in[7];
    const float* b3       = (const float*)d_in[8];
    const float* gamma4   = (const float*)d_in[9];
    const float* beta4    = (const float*)d_in[10];
    const float* w4       = (const float*)d_in[11];
    const float* b4       = (const float*)d_in[12];
    const float* gamma5   = (const float*)d_in[13];
    const float* beta5    = (const float*)d_in[14];
    const float* w5       = (const float*)d_in[15];
    const float* b5       = (const float*)d_in[16];

    char* ws = (char*)d_ws;
    size_t off = 0;
    auto carve = [&](size_t bytes) { char* p = ws + off; off = (off + bytes + 255) & ~(size_t)255; return p; };
    __bf16* Xr         = (__bf16*)carve((size_t)ROWS * 40 * 2);   // 2.62 MB
    float*  inputs_raw = (float*) carve((size_t)BATCH * DIN * 4); // 6.79 MB
    __bf16* NH         = (__bf16*)carve((size_t)ROWS * 64 * 2);   // 4.19 MB
    __bf16* W0bf       = (__bf16*)carve((size_t)K1P * 128 * 2);
    __bf16* W1bf       = (__bf16*)carve((size_t)K2R * 128 * 2);
    float*  z3         = (float*) carve((size_t)BATCH * 64 * 4);
    float*  h3         = (float*) carve((size_t)BATCH * 64 * 4);
    float*  z4         = (float*) carve((size_t)BATCH * 32 * 4);
    float*  h4         = (float*) carve((size_t)BATCH * 32 * 4);

    float* out_final = (float*)d_out;            // [0 .. B)
    float* inputs_n  = out_final + BATCH;        // [B .. B + B*DIN): normalized inputs

    // 1) pack weights to bf16 (zero-padded K)
    wconv_kernel<<<(K1P * 128 + 255) / 256, 256, 0, stream>>>(W0, W0bf, K1R * 128, K1P * 128);
    wconv_kernel<<<(K2R * 128 + 255) / 256, 256, 0, stream>>>(W1, W1bf, K2R * 128, K2R * 128);

    // 2) embedding gather + transpose + inputs_raw[:, 0..637)
    gather_kernel<<<(BATCH * F0 + 255) / 256, 256, 0, stream>>>(in_hash, table, input_x, Xr, inputs_raw);

    // 3) cross GEMM 1 (WMMA bf16, async double-buffered W): next_h + direct0
    cross_gemm<1><<<ROWS / 128, 256, CL_OFF, stream>>>(
        Xr, nullptr, W0bf, NH, inputs_raw, K1P, K1R);

    // 4) cross GEMM 2 (WMMA bf16, async double-buffered W): fused direct1
    cross_gemm<2><<<ROWS / 128, 256, CL_OFF + 128 * 64 * 2, stream>>>(
        Xr, NH, W1bf, nullptr, inputs_raw, K2R, K2R);

    // 5) BN1 -> normalized `inputs` (second output), also feeds the MLP
    bn_kernel<<<(DIN + 31) / 32, 256, 0, stream>>>(inputs_raw, inputs_n, gamma1, beta1, DIN, 0);

    // 6) MLP tail
    mlp1_kernel<<<(BATCH * 64) / 256, 256, 0, stream>>>(inputs_n, w3, b3, z3);
    bn_kernel<<<2, 256, 0, stream>>>(z3, h3, gamma4, beta4, 64, 1);
    mlp2_kernel<<<(BATCH * 32) / 256, 256, 0, stream>>>(h3, w4, b4, z4);
    bn_kernel<<<1, 256, 0, stream>>>(z4, h4, gamma5, beta5, 32, 1);
    final_kernel<<<(BATCH + 255) / 256, 256, 0, stream>>>(h4, w5, b5, out_final);
}
